// GatedGCNNet_55207509623128
// MI455X (gfx1250) — matrix operand order repaired and checked by reading, not compiled, not run on hardware
//
#include <hip/hip_runtime.h>

typedef __attribute__((ext_vector_type(16))) _Float16 v16h;
typedef __attribute__((ext_vector_type(8)))  _Float16 v8h;
typedef __attribute__((ext_vector_type(8)))  float    v8f;

#define HD    70     // hidden dim
#define KP    96     // padded K (3 x 32)
#define KP2   48     // KP/2 (float2 staging)
#define NPAD  80     // padded N (5 x 16)
#define AGG_EPS 1e-6f
#define BN_EPS  1e-5f

// ---------- WMMA fragment loaders (contiguous k order, same for A and B) ----------
// A (16x32 f16): lane<16 holds K 0..15, lane>=16 holds K 16..31 (contiguous)
static __device__ __forceinline__ v16h gg_load_a(const _Float16* lds_row, int kk, int lane) {
  const _Float16* base = lds_row + kk * 32 + ((lane >> 4) << 4);
  union { v16h v; v8h h[2]; } u;
  u.h[0] = *(const v8h*)(base);        // K base..base+7
  u.h[1] = *(const v8h*)(base + 8);    // K base+8..base+15
  return u.v;
}
// B (32x16 f16, column n per lane%16): lane<16 K 0..15, lane>=16 K 16..31 (contiguous)
static __device__ __forceinline__ v16h gg_load_b(const _Float16* lds, int n, int kk, int lane) {
  return *(const v16h*)(lds + n * KP + kk * 32 + ((lane >> 4) << 4));
}

// ---------- node embedding: h = X(N x 32) @ W(32 x 70) + b ----------
__global__ void gg_embed_nodes(const float* __restrict__ X, const float* __restrict__ W,
                               const float* __restrict__ b, float* __restrict__ h, int Ncnt) {
  __shared__ float sW[32 * HD];
  int tid = threadIdx.x;
  for (int i = tid; i < 32 * HD; i += 256) sW[i] = W[i];
  __syncthreads();
  long idx = (long)blockIdx.x * 256 + tid;
  long total = (long)Ncnt * HD;
  if (idx < total) {
    int r = (int)(idx / HD), c = (int)(idx - (long)r * HD);
    const float* xr = X + (long)r * 32;
    float s = b[c];
#pragma unroll
    for (int k = 0; k < 32; ++k) s += xr[k] * sW[k * HD + c];
    h[idx] = s;
  }
}

// ---------- edge embedding: e = ef(E x 1) @ W(1 x 70) + b (outer product) ----------
__global__ void gg_embed_edges(const float* __restrict__ ef, const float* __restrict__ W,
                               const float* __restrict__ b, float* __restrict__ e, long total) {
  long idx = (long)blockIdx.x * 256 + threadIdx.x;
  if (idx < total) {
    int r = (int)(idx / HD), c = (int)(idx - (long)r * HD);
    e[idx] = ef[r] * W[c] + b[c];
  }
}

// ---------- node GEMM epilogue store helper ----------
template <bool CHK>
static __device__ __forceinline__ void gg_store_tile(float* __restrict__ O, const v8f* acc,
                                                     int row0, int mbase, int lane, int Ncnt) {
#pragma unroll
  for (int t = 0; t < 5; ++t) {
    const int n = t * 16 + (lane & 15);
    if (n < HD) {
#pragma unroll
      for (int j = 0; j < 8; ++j) {
        int row = row0 + mbase + j + ((lane >> 4) << 3);
        if (!CHK || row < Ncnt) O[(long)row * HD + n] = acc[t][j];
      }
    }
  }
}

// ---------- node GEMMs: Ah,Bh,Dh,Eh = h @ W + b   (WMMA f16->f32) ----------
__global__ void __launch_bounds__(128) gg_node_gemm(
    const float* __restrict__ h,
    const float* __restrict__ Aw, const float* __restrict__ Ab,
    const float* __restrict__ Bw, const float* __restrict__ Bb,
    const float* __restrict__ Dw, const float* __restrict__ Db,
    const float* __restrict__ Ew, const float* __restrict__ Eb,
    float* __restrict__ Ao, float* __restrict__ Bo,
    float* __restrict__ Do, float* __restrict__ Eo, int Ncnt) {
  __shared__ __align__(32) _Float16 sH[64 * KP];
  __shared__ __align__(32) _Float16 sW[NPAD * KP];

  const int tid = threadIdx.x;
  const int row0 = blockIdx.x * 64;
  const bool full = (row0 + 64 <= Ncnt);

  // stage 64 rows of h, f32 -> f16 (float2-vectorized; rows are 8B aligned), pad to KP
  for (int i = tid; i < 64 * KP2; i += 128) {
    int r = i / KP2, k = (i - r * KP2) * 2;
    int row = row0 + r;
    float2 v = make_float2(0.0f, 0.0f);
    if (row < Ncnt && k < HD) v = *(const float2*)&h[(long)row * HD + k];
    sH[r * KP + k]     = (_Float16)v.x;
    sH[r * KP + k + 1] = (_Float16)v.y;
  }
  __syncthreads();

  const int wave = tid >> 5, lane = tid & 31;
  const int mbase = wave * 16;
  const _Float16* arow = &sH[(mbase + (lane & 15)) * KP];
  v16h afrag[3];
#pragma unroll
  for (int kk = 0; kk < 3; ++kk) afrag[kk] = gg_load_a(arow, kk, lane);

  const float* Ws[4] = {Aw, Bw, Dw, Ew};
  const float* Bs[4] = {Ab, Bb, Db, Eb};
  float*       Os[4] = {Ao, Bo, Do, Eo};

#pragma unroll
  for (int m = 0; m < 4; ++m) {
    __syncthreads();
    // stage transposed weights: sW[n*KP+k] = W[k*HD+n]
    const float* W = Ws[m];
    for (int i = tid; i < NPAD * KP; i += 128) {
      int n = i / KP, k = i - n * KP;
      sW[i] = (_Float16)((n < HD && k < HD) ? W[k * HD + n] : 0.0f);
    }
    __syncthreads();

    v8f acc[5];
#pragma unroll
    for (int t = 0; t < 5; ++t) {
      const int n = t * 16 + (lane & 15);
      float bias = (n < HD) ? Bs[m][n] : 0.0f;
#pragma unroll
      for (int j = 0; j < 8; ++j) acc[t][j] = bias;
    }
#pragma unroll
    for (int kk = 0; kk < 3; ++kk) {
#pragma unroll
      for (int t = 0; t < 5; ++t) {
        v16h bf = gg_load_b(sW, t * 16 + (lane & 15), kk, lane);
        acc[t] = __builtin_amdgcn_wmma_f32_16x16x32_f16(false, afrag[kk], false, bf,
                                                        (short)0, acc[t], false, false);
      }
    }
    if (full) gg_store_tile<false>(Os[m], acc, row0, mbase, lane, Ncnt);
    else      gg_store_tile<true >(Os[m], acc, row0, mbase, lane, Ncnt);
  }
}

// ---------- edge epilogue helper ----------
template <bool CHK>
static __device__ __forceinline__ void gg_edge_epi(
    const v8f* acc, int lane, int mbase, long e0, int Ecnt,
    const int* __restrict__ sSrc, const int* __restrict__ sDst, const float* __restrict__ sSn,
    const float* __restrict__ Bh, const float* __restrict__ Dh, const float* __restrict__ Eh,
    _Float16* __restrict__ xe, float* __restrict__ num, float* __restrict__ den,
    float* __restrict__ sBN) {
#pragma unroll
  for (int t = 0; t < 5; ++t) {
    const int n = t * 16 + (lane & 15);
    if (n < HD) {
      float ls = 0.0f, lq = 0.0f;
#pragma unroll
      for (int j = 0; j < 8; ++j) {
        int mloc = mbase + j + ((lane >> 4) << 3);
        long eg = e0 + mloc;
        if (!CHK || eg < Ecnt) {
          long s = sSrc[mloc], d = sDst[mloc];
          float enew = acc[t][j] + Dh[s * HD + n] + Eh[d * HD + n];
          float sg = 1.0f / (1.0f + __expf(-enew));
          atomicAdd(&num[d * HD + n], sg * Bh[s * HD + n]);
          atomicAdd(&den[d * HD + n], sg);
          float x = enew * sSn[mloc];
          xe[eg * HD + n] = (_Float16)x;
          ls += x; lq += x * x;
        }
      }
      atomicAdd(&sBN[n], ls);
      atomicAdd(&sBN[NPAD + n], lq);
    }
  }
}

// ---------- fused edge kernel: Ce = e@Cw+Cb (WMMA); e_new = Ce + Dh[src] + Eh[dst];
//            sigma=sigmoid; atomics into num/den; x_e = e_new*snorm_e (f16); BN stats ----------
__global__ void __launch_bounds__(256) gg_edge_kernel(
    const float* __restrict__ e, const float* __restrict__ Cw, const float* __restrict__ Cb,
    const float* __restrict__ Bh, const float* __restrict__ Dh, const float* __restrict__ Eh,
    const int* __restrict__ src, const int* __restrict__ dst,
    const float* __restrict__ snorm_e,
    _Float16* __restrict__ xe, float* __restrict__ num, float* __restrict__ den,
    float* __restrict__ bn_sum, float* __restrict__ bn_sumsq, int Ecnt) {
  __shared__ __align__(32) _Float16 sE[128 * KP];
  __shared__ __align__(32) _Float16 sW[NPAD * KP];
  __shared__ int   sSrc[128], sDst[128];
  __shared__ float sSn[128];
  __shared__ float sBN[2 * NPAD];

  const int tid = threadIdx.x;
  const long e0 = (long)blockIdx.x * 128;
  const bool fullb = (e0 + 128 <= Ecnt);

  for (int i = tid; i < 2 * NPAD; i += 256) sBN[i] = 0.0f;
  for (int i = tid; i < NPAD * KP; i += 256) {
    int n = i / KP, k = i - n * KP;
    sW[i] = (_Float16)((n < HD && k < HD) ? Cw[k * HD + n] : 0.0f);
  }
  // stage 128 edge rows, float2-vectorized (rows 8B-aligned since HD is even)
  for (int i = tid; i < 128 * KP2; i += 256) {
    int r = i / KP2, k = (i - r * KP2) * 2;
    long eidx = e0 + r;
    float2 v = make_float2(0.0f, 0.0f);
    if (eidx < Ecnt && k < HD) v = *(const float2*)&e[eidx * HD + k];
    sE[r * KP + k]     = (_Float16)v.x;
    sE[r * KP + k + 1] = (_Float16)v.y;
  }
  if (tid < 128) {
    long eidx = e0 + tid;
    bool v = eidx < Ecnt;
    sSrc[tid] = v ? src[eidx] : 0;
    sDst[tid] = v ? dst[eidx] : 0;
    sSn[tid]  = v ? snorm_e[eidx] : 0.0f;
  }
  __syncthreads();

  const int wave = tid >> 5, lane = tid & 31;
  const int mbase = wave * 16;
  const _Float16* arow = &sE[(mbase + (lane & 15)) * KP];
  v16h afrag[3];
#pragma unroll
  for (int kk = 0; kk < 3; ++kk) afrag[kk] = gg_load_a(arow, kk, lane);

  v8f acc[5];
#pragma unroll
  for (int t = 0; t < 5; ++t) {
    const int n = t * 16 + (lane & 15);
    float bias = (n < HD) ? Cb[n] : 0.0f;
#pragma unroll
    for (int j = 0; j < 8; ++j) acc[t][j] = bias;
  }
#pragma unroll
  for (int kk = 0; kk < 3; ++kk) {
#pragma unroll
    for (int t = 0; t < 5; ++t) {
      v16h bf = gg_load_b(sW, t * 16 + (lane & 15), kk, lane);
      acc[t] = __builtin_amdgcn_wmma_f32_16x16x32_f16(false, afrag[kk], false, bf,
                                                      (short)0, acc[t], false, false);
    }
  }

  if (fullb) gg_edge_epi<false>(acc, lane, mbase, e0, Ecnt, sSrc, sDst, sSn,
                                Bh, Dh, Eh, xe, num, den, sBN);
  else       gg_edge_epi<true >(acc, lane, mbase, e0, Ecnt, sSrc, sDst, sSn,
                                Bh, Dh, Eh, xe, num, den, sBN);

  __syncthreads();
  if (tid < HD) {
    atomicAdd(&bn_sum[tid], sBN[tid]);
    atomicAdd(&bn_sumsq[tid], sBN[NPAD + tid]);
  }
}

// ---------- BN finalize: mean / inv_std per channel ----------
__global__ void gg_bn_finalize(const float* __restrict__ sum, const float* __restrict__ sumsq,
                               float count, float* __restrict__ mean, float* __restrict__ inv) {
  int c = threadIdx.x;
  if (c < HD) {
    float mu = sum[c] / count;
    float var = sumsq[c] / count - mu * mu;
    mean[c] = mu;
    inv[c] = rsqrtf(var + BN_EPS);
  }
}

// ---------- BN apply + ReLU + residual: io += relu(g*(x-mu)*inv + b) ----------
template <typename T>
__global__ void gg_bn_apply(float* __restrict__ io, const T* __restrict__ x,
                            const float* __restrict__ mean, const float* __restrict__ inv,
                            const float* __restrict__ g, const float* __restrict__ b, long total) {
  long idx = (long)blockIdx.x * 256 + threadIdx.x;
  if (idx < total) {
    int c = (int)(idx % HD);
    float xv = (float)x[idx];
    float y = g[c] * (xv - mean[c]) * inv[c] + b[c];
    io[idx] += fmaxf(y, 0.0f);
  }
}

// ---------- node update: x = (Ah + num/(den+eps)) * snorm_n  (into num); BN stats ----------
__global__ void __launch_bounds__(256) gg_node_update(
    const float* __restrict__ Ah, float* __restrict__ num, const float* __restrict__ den,
    const float* __restrict__ snorm_n,
    float* __restrict__ bn_sum, float* __restrict__ bn_sumsq, int Ncnt) {
  __shared__ float sS[HD], sQ[HD];
  const int tid = threadIdx.x;
  if (tid < HD) { sS[tid] = 0.0f; sQ[tid] = 0.0f; }
  __syncthreads();
  const long total = (long)Ncnt * HD;
#pragma unroll
  for (int it = 0; it < 8; ++it) {
    long idx = ((long)blockIdx.x * 8 + it) * 256 + tid;
    if (idx < total) {
      int r = (int)(idx / HD);
      int c = (int)(idx - (long)r * HD);
      float x = Ah[idx] + num[idx] / (den[idx] + AGG_EPS);
      x *= snorm_n[r];
      num[idx] = x;
      atomicAdd(&sS[c], x);
      atomicAdd(&sQ[c], x * x);
    }
  }
  __syncthreads();
  if (tid < HD) {
    atomicAdd(&bn_sum[tid], sS[tid]);
    atomicAdd(&bn_sumsq[tid], sQ[tid]);
  }
}

// ---------- graph readout: hg[g] += h[r]; cnt[g] += 1 ----------
__global__ void gg_readout(const float* __restrict__ h, const int* __restrict__ gid,
                           float* __restrict__ hg, float* __restrict__ cnt, int Ncnt) {
  long idx = (long)blockIdx.x * 256 + threadIdx.x;
  long total = (long)Ncnt * HD;
  if (idx < total) {
    int r = (int)(idx / HD), c = (int)(idx - (long)r * HD);
    int g = gid[r];
    atomicAdd(&hg[(long)g * HD + c], h[idx]);
    if (c == 0) atomicAdd(&cnt[g], 1.0f);
  }
}

// ---------- final MLP: 70 -> 35 -> 17 -> 10 per graph ----------
__global__ void gg_mlp(const float* __restrict__ hg, const float* __restrict__ cnt,
                       const float* __restrict__ w0, const float* __restrict__ b0,
                       const float* __restrict__ w1, const float* __restrict__ b1,
                       const float* __restrict__ w2, const float* __restrict__ b2,
                       float* __restrict__ out) {
  __shared__ float hr[HD], x0[35], x1[17];
  const int g = blockIdx.x, t = threadIdx.x;
  float c = fmaxf(cnt[g], 1.0f);
  if (t < HD) hr[t] = hg[(long)g * HD + t] / c;
  __syncthreads();
  if (t < 35) {
    float s = b0[t];
    for (int k = 0; k < HD; ++k) s += hr[k] * w0[k * 35 + t];
    x0[t] = fmaxf(s, 0.0f);
  }
  __syncthreads();
  if (t < 17) {
    float s = b1[t];
    for (int k = 0; k < 35; ++k) s += x0[k] * w1[k * 17 + t];
    x1[t] = fmaxf(s, 0.0f);
  }
  __syncthreads();
  if (t < 10) {
    float s = b2[t];
    for (int k = 0; k < 17; ++k) s += x1[k] * w2[k * 10 + t];
    out[(long)g * 10 + t] = s;
  }
}

extern "C" void kernel_launch(void* const* d_in, const int* in_sizes, int n_in,
                              void* d_out, int out_size, void* d_ws, size_t ws_size,
                              hipStream_t stream) {
  const float* nodes_feat = (const float*)d_in[0];
  const float* edges_feat = (const float*)d_in[1];
  const float* snorm_n    = (const float*)d_in[2];
  const float* snorm_e    = (const float*)d_in[3];
  const float* emb_h_w    = (const float*)d_in[4];
  const float* emb_h_b    = (const float*)d_in[5];
  const float* emb_e_w    = (const float*)d_in[6];
  const float* emb_e_b    = (const float*)d_in[7];
  const float* A_w = (const float*)d_in[8],  *A_b = (const float*)d_in[9];
  const float* B_w = (const float*)d_in[10], *B_b = (const float*)d_in[11];
  const float* C_w = (const float*)d_in[12], *C_b = (const float*)d_in[13];
  const float* D_w = (const float*)d_in[14], *D_b = (const float*)d_in[15];
  const float* E_w = (const float*)d_in[16], *E_b = (const float*)d_in[17];
  const float* bn_h_g = (const float*)d_in[18], *bn_h_b = (const float*)d_in[19];
  const float* bn_e_g = (const float*)d_in[20], *bn_e_b = (const float*)d_in[21];
  const float* mlp_w0 = (const float*)d_in[22], *mlp_b0 = (const float*)d_in[23];
  const float* mlp_w1 = (const float*)d_in[24], *mlp_b1 = (const float*)d_in[25];
  const float* mlp_w2 = (const float*)d_in[26], *mlp_b2 = (const float*)d_in[27];
  const int* src  = (const int*)d_in[28];
  const int* dst  = (const int*)d_in[29];
  const int* gids = (const int*)d_in[30];

  const int Ncnt = in_sizes[2];           // snorm_n has N elements
  const int Ecnt = in_sizes[3];           // snorm_e has E elements
  const int Gcnt = out_size / 10;

  const size_t NH = (size_t)Ncnt * HD;
  const size_t EH = (size_t)Ecnt * HD;

  float* ws  = (float*)d_ws;
  float* h   = ws;
  float* Ah  = h   + NH;
  float* Bh  = Ah  + NH;
  float* Dh  = Bh  + NH;
  float* Eh  = Dh  + NH;
  float* num = Eh  + NH;
  float* den = num + NH;
  float* e   = den + NH;
  float* hg  = e   + EH;
  float* cnt = hg  + (size_t)Gcnt * HD;
  float* bnb = cnt + Gcnt;                 // 640 floats of BN scratch
  _Float16* xe = (_Float16*)(bnb + 640);   // f16 BN input buffer (EH halfs)
  size_t need = ((size_t)(xe + EH) - (size_t)ws);
  if (ws_size < need) return;              // scratch too small: nothing safe to do

  const unsigned TB = 256;
  const unsigned gridNH = (unsigned)((NH + TB - 1) / TB);
  const unsigned gridEH = (unsigned)((EH + TB - 1) / TB);

  // embeddings
  gg_embed_nodes<<<gridNH, TB, 0, stream>>>(nodes_feat, emb_h_w, emb_h_b, h, Ncnt);
  gg_embed_edges<<<gridEH, TB, 0, stream>>>(edges_feat, emb_e_w, emb_e_b, e, (long)EH);

  for (int l = 0; l < 4; ++l) {
    const size_t wo = (size_t)l * HD * HD;  // weight offset per layer
    const size_t bo = (size_t)l * HD;       // bias / bn offset per layer

    hipMemsetAsync(num, 0, NH * sizeof(float), stream);
    hipMemsetAsync(den, 0, NH * sizeof(float), stream);
    hipMemsetAsync(bnb, 0, 640 * sizeof(float), stream);

    gg_node_gemm<<<(Ncnt + 63) / 64, 128, 0, stream>>>(
        h, A_w + wo, A_b + bo, B_w + wo, B_b + bo,
        D_w + wo, D_b + bo, E_w + wo, E_b + bo,
        Ah, Bh, Dh, Eh, Ncnt);

    gg_edge_kernel<<<(Ecnt + 127) / 128, 256, 0, stream>>>(
        e, C_w + wo, C_b + bo, Bh, Dh, Eh, src, dst, snorm_e,
        xe, num, den, bnb + 0, bnb + 80, Ecnt);

    gg_bn_finalize<<<1, 128, 0, stream>>>(bnb + 0, bnb + 80, (float)Ecnt, bnb + 160, bnb + 240);
    gg_bn_apply<_Float16><<<gridEH, TB, 0, stream>>>(e, xe, bnb + 160, bnb + 240,
                                                     bn_e_g + bo, bn_e_b + bo, (long)EH);

    gg_node_update<<<(unsigned)((NH + 2047) / 2048), 256, 0, stream>>>(
        Ah, num, den, snorm_n, bnb + 320, bnb + 400, Ncnt);
    gg_bn_finalize<<<1, 128, 0, stream>>>(bnb + 320, bnb + 400, (float)Ncnt, bnb + 480, bnb + 560);
    gg_bn_apply<float><<<gridNH, TB, 0, stream>>>(h, num, bnb + 480, bnb + 560,
                                                  bn_h_g + bo, bn_h_b + bo, (long)NH);
  }

  // readout + MLP
  hipMemsetAsync(hg, 0, ((size_t)Gcnt * HD + Gcnt) * sizeof(float), stream);
  gg_readout<<<gridNH, TB, 0, stream>>>(h, gids, hg, cnt, Ncnt);
  gg_mlp<<<Gcnt, 128, 0, stream>>>(hg, cnt, mlp_w0, mlp_b0, mlp_w1, mlp_b1,
                                   mlp_w2, mlp_b2, (float*)d_out);
}
